// QuantLinear_30923764532091
// MI455X (gfx1250) — 8-bit matrix operand order repaired and checked by reading, compiled, not run on hardware
//
#include <hip/hip_runtime.h>
#include <cstdint>

typedef int v8i __attribute__((ext_vector_type(8)));
typedef int gvec4 __attribute__((vector_size(16)));
typedef __attribute__((address_space(1))) gvec4 gvec4_as1;
typedef __attribute__((address_space(3))) gvec4 gvec4_as3;

#define M_DIM 16384
#define N_DIM 2048
#define K_DIM 2048
#define GS    128
#define NG    (K_DIM / GS)
#define PITCH 144

#if __has_builtin(__builtin_amdgcn_global_load_async_to_lds_b128)
#define HAVE_ASYNC_LDS 1
#endif

__device__ __forceinline__ void cp_async16(void* lds, const void* gsrc) {
#ifdef HAVE_ASYNC_LDS
    __builtin_amdgcn_global_load_async_to_lds_b128(
        (gvec4_as1*)(uintptr_t)gsrc,
        (gvec4_as3*)(uint32_t)(uintptr_t)lds,
        0, 0);
#else
    *(int4*)lds = *(const int4*)gsrc;
#endif
}

__device__ __forceinline__ void async_wait_all() {
#ifdef HAVE_ASYNC_LDS
#if __has_builtin(__builtin_amdgcn_s_wait_asynccnt)
    __builtin_amdgcn_s_wait_asynccnt(0);
#else
    asm volatile("s_wait_asynccnt 0x0" ::: "memory");
#endif
#endif
}

__global__ __launch_bounds__(256) void quant_kernel(const float* __restrict__ src,
                                                    char* __restrict__ qout,
                                                    float* __restrict__ sout,
                                                    int n_groups) {
    const int wave = threadIdx.x >> 5;
    const int lane = threadIdx.x & 31;
    const int gid  = blockIdx.x * 8 + wave;
    if (gid >= n_groups) return;

    const int row = gid >> 4;
    const int g   = gid & (NG - 1);

    const float* p = src + (size_t)row * K_DIM + (size_t)g * GS + (size_t)lane * 4;
    float4 v = *(const float4*)p;

    float amax = fmaxf(fmaxf(fabsf(v.x), fabsf(v.y)), fmaxf(fabsf(v.z), fabsf(v.w)));
#pragma unroll
    for (int off = 16; off > 0; off >>= 1)
        amax = fmaxf(amax, __shfl_xor(amax, off, 32));

    const float scale = fmaxf(amax * (1.0f / 127.0f), 1e-8f);
    const float inv   = 1.0f / scale;

    int q0 = (int)rintf(v.x * inv);
    int q1 = (int)rintf(v.y * inv);
    int q2 = (int)rintf(v.z * inv);
    int q3 = (int)rintf(v.w * inv);
    q0 = max(-128, min(127, q0));
    q1 = max(-128, min(127, q1));
    q2 = max(-128, min(127, q2));
    q3 = max(-128, min(127, q3));

    unsigned int packed = (unsigned)(q0 & 255) | ((unsigned)(q1 & 255) << 8) |
                          ((unsigned)(q2 & 255) << 16) | ((unsigned)(q3 & 255) << 24);

    ((unsigned int*)qout)[((size_t)row * K_DIM + (size_t)g * GS) / 4 + lane] = packed;
    if (lane == 0) sout[(size_t)row * NG + g] = scale;
}

__global__ __launch_bounds__(256) void gemm_kernel(const char* __restrict__ qx,
                                                   const char* __restrict__ qw,
                                                   const float* __restrict__ sx,
                                                   const float* __restrict__ sw,
                                                   float* __restrict__ out) {
    __shared__ char  lsA[2][128 * PITCH];
    __shared__ char  lsB[2][128 * PITCH];
    __shared__ float lsx[NG * 128];
    __shared__ float lsw[NG * 128];

    const int tid = threadIdx.x;
    const int bm  = blockIdx.x * 128;
    const int bn  = blockIdx.y * 128;

#pragma unroll
    for (int i = tid; i < NG * 128; i += 256) {
        const int g = i >> 7, r = i & 127;
        lsx[i] = sx[(size_t)(bm + r) * NG + g];
        lsw[i] = sw[(size_t)(bn + r) * NG + g];
    }

#pragma unroll
    for (int p = 0; p < 4; ++p) {
        const int idx = p * 256 + tid;
        const int row = idx >> 3;
        const int c   = (idx & 7) * 16;
        cp_async16(&lsA[0][row * PITCH + c], qx + (size_t)(bm + row) * K_DIM + c);
        cp_async16(&lsB[0][row * PITCH + c], qw + (size_t)(bn + row) * K_DIM + c);
    }
    async_wait_all();
    __syncthreads();

    const int wave = tid >> 5;
    const int lane = tid & 31;
    const int wm   = wave & 1;
    const int wn   = wave >> 1;
    const int half = lane >> 4;
    const int l16  = lane & 15;

    float acc[4][2][8];
#pragma unroll
    for (int mt = 0; mt < 4; ++mt)
#pragma unroll
        for (int nt = 0; nt < 2; ++nt)
#pragma unroll
            for (int r = 0; r < 8; ++r) acc[mt][nt][r] = 0.0f;

    for (int g = 0; g < NG; ++g) {
        const int cur = g & 1;

        if (g + 1 < NG) {
            const int nxt = cur ^ 1;
            const int k1  = (g + 1) * GS;
#pragma unroll
            for (int p = 0; p < 4; ++p) {
                const int idx = p * 256 + tid;
                const int row = idx >> 3;
                const int c   = (idx & 7) * 16;
                cp_async16(&lsA[nxt][row * PITCH + c],
                           qx + (size_t)(bm + row) * K_DIM + k1 + c);
                cp_async16(&lsB[nxt][row * PITCH + c],
                           qw + (size_t)(bn + row) * K_DIM + k1 + c);
            }
        }

        const char* As = &lsA[cur][0];
        const char* Bs = &lsB[cur][0];

        v8i Bf[2][2];
#pragma unroll
        for (int nt = 0; nt < 2; ++nt) {
            const char* bp = Bs + (wn * 32 + nt * 16 + l16) * PITCH + half * 16;
#pragma unroll
            for (int kh = 0; kh < 2; ++kh) {
                int4 b03 = *(const int4*)(bp + kh * 64);
                int4 b47 = *(const int4*)(bp + kh * 64 + 32);
                v8i b;
                b[0] = b03.x; b[1] = b03.y; b[2] = b03.z; b[3] = b03.w;
                b[4] = b47.x; b[5] = b47.y; b[6] = b47.z; b[7] = b47.w;
                Bf[nt][kh] = b;
            }
        }

        const float* sxg = &lsx[g * 128];
        const float* swg = &lsw[g * 128];

#pragma unroll
        for (int mt = 0; mt < 4; ++mt) {
            const char* ap = As + (wm * 64 + mt * 16 + l16) * PITCH + half * 16;
            v8i Af[2];
#pragma unroll
            for (int kh = 0; kh < 2; ++kh) {
                int2 a01 = *(const int2*)(ap + kh * 64 + 0);
                int2 a23 = *(const int2*)(ap + kh * 64 + 8);
                int2 a45 = *(const int2*)(ap + kh * 64 + 32);
                int2 a67 = *(const int2*)(ap + kh * 64 + 40);
                v8i a;
                a[0] = a01.x; a[1] = a01.y; a[2] = a23.x; a[3] = a23.y;
                a[4] = a45.x; a[5] = a45.y; a[6] = a67.x; a[7] = a67.y;
                Af[kh] = a;
            }

            const int mloc = wm * 64 + mt * 16 + half * 8;
            float4 s03 = *(const float4*)(sxg + mloc);
            float4 s47 = *(const float4*)(sxg + mloc + 4);
            const float sxv[8] = {s03.x, s03.y, s03.z, s03.w,
                                  s47.x, s47.y, s47.z, s47.w};

#pragma unroll
            for (int nt = 0; nt < 2; ++nt) {
                const float swv = swg[wn * 32 + nt * 16 + l16];
                v8i ci = {};
                ci = __builtin_amdgcn_wmma_i32_16x16x64_iu8(
                        true, Af[0], true, Bf[nt][0], ci, false, false);
                ci = __builtin_amdgcn_wmma_i32_16x16x64_iu8(
                        true, Af[1], true, Bf[nt][1], ci, false, false);
#pragma unroll
                for (int r = 0; r < 8; ++r)
                    acc[mt][nt][r] = fmaf((float)ci[r], sxv[r] * swv,
                                          acc[mt][nt][r]);
            }
        }

        async_wait_all();
        __syncthreads();
    }

    const int m_wave = bm + wm * 64;
    const int n_wave = bn + wn * 32;
#pragma unroll
    for (int mt = 0; mt < 4; ++mt)
#pragma unroll
        for (int nt = 0; nt < 2; ++nt)
#pragma unroll
            for (int r = 0; r < 8; ++r) {
                const int m = m_wave + mt * 16 + half * 8 + r;
                const int n = n_wave + nt * 16 + l16;
                out[(size_t)m * N_DIM + n] = acc[mt][nt][r];
            }
}

extern "C" void kernel_launch(void* const* d_in, const int* in_sizes, int n_in,
                              void* d_out, int out_size, void* d_ws, size_t ws_size,
                              hipStream_t stream) {
    const float* x = (const float*)d_in[0];
    const float* w = (const float*)d_in[1];
    float* out = (float*)d_out;

    char* base = (char*)d_ws;
    char*  qx = base;
    char*  qw = qx + (size_t)M_DIM * K_DIM;
    float* sx = (float*)(qw + (size_t)N_DIM * K_DIM);
    float* sw = sx + (size_t)M_DIM * NG;

    quant_kernel<<<(M_DIM * NG) / 8, 256, 0, stream>>>(x, qx, sx, M_DIM * NG);
    quant_kernel<<<(N_DIM * NG) / 8, 256, 0, stream>>>(w, qw, sw, N_DIM * NG);

    dim3 grid(M_DIM / 128, N_DIM / 128);
    gemm_kernel<<<grid, 256, 0, stream>>>(qx, qw, sx, sw, out);
}
